// BinaryTreeLogicNet_19894288515463
// MI455X (gfx1250) — matrix operand order repaired and checked by reading, not compiled, not run on hardware
//
#include <hip/hip_runtime.h>

typedef __attribute__((ext_vector_type(16))) __bf16 v16bf;
typedef __attribute__((ext_vector_type(8)))  float  v8f;
typedef __attribute__((ext_vector_type(4)))  float  v4f;
typedef __attribute__((ext_vector_type(4)))  unsigned int v4u;

#define LCOLS 256
#define EPSV 1e-6f
#define BIAS_SHIFT_V (-2.0f)

// LDS layout: W tile, 256 rows, padded pitch to avoid 16-way bank conflicts.
// Row = 512B of bf16 data; pitch 528B => bank stride 132 % 64 = 4 banks/row.
#define W_PITCH_B   528
#define W_LDS_BYTES (256 * W_PITCH_B)          // 135168
#define NODE_OFF    W_LDS_BYTES                // packed [w0,w1,b] * 255 floats
#define SMEM_BYTES  (W_LDS_BYTES + 255 * 3 * 4 + 64)

// Fast sigmoid: v_exp_f32 + v_add + v_rcp_f32.
__device__ __forceinline__ float sigmoidf_(float z) {
    return __builtin_amdgcn_rcpf(1.0f + __expf(-z));
}

// lam*min + (1-lam)*max == fma(lam, min-max, max)
__device__ __forceinline__ float soft_gcd(float lv, float rv, float lam) {
    float a  = fabsf(lv) + EPSV;
    float b  = fabsf(rv) + EPSV;
    float mn = fminf(a, b);
    float mx = fmaxf(a, b);
    return fmaf(lam, mn - mx, mx);
}

// One-time conversion of W_leaf (256x256 f32, row-major) -> bf16 in workspace.
__global__ void w_to_bf16(const float* __restrict__ W, __bf16* __restrict__ Wb) {
    int i = blockIdx.x * blockDim.x + threadIdx.x;
    Wb[i] = (__bf16)W[i];
}

// One wave = 16 rows. GEMM via v_wmma_f32_16x16x32_bf16 with B staged in LDS,
// tree via shfl_xor butterflies (no LDS memory traffic for the tree).
__global__ __launch_bounds__(256) void tree_logic_net(
    const float* __restrict__ x,       // (B, 256) f32
    const __bf16* __restrict__ Wb,     // (256, 256) bf16, row-major
    const float* __restrict__ weights, // (255, 2)
    const float* __restrict__ biases,  // (255,)
    const float* __restrict__ w_out,   // (1,1)
    const float* __restrict__ b_out,   // (1,)
    float* __restrict__ out)           // (B,)
{
    extern __shared__ char smem[];
    float* nodes = (float*)(smem + NODE_OFF);

    const int tid  = threadIdx.x;
    const int lane = tid & 31;
    const int wave = tid >> 5;
    const int g    = lane >> 4;   // K-half selector (A/B fragment layout)
    const int n    = lane & 15;   // row (A) / column (B,C) within tile
    const int row0 = (blockIdx.x * 8 + wave) * 16;

    // ---- Stage W (128 KB bf16) into padded LDS, coalesced 16B chunks ----
    {
        const v4u* Wg = (const v4u*)Wb;   // 8192 x 16B chunks, 32 chunks/row
        #pragma unroll
        for (int i = 0; i < 32; ++i) {
            int m    = tid + 256 * i;
            int row  = m >> 5;
            int colc = m & 31;
            *(v4u*)(smem + row * W_PITCH_B + colc * 16) = Wg[m];
        }
        // Tree node params: [w0, w1, bias] per node.
        if (tid < 255) {
            nodes[3 * tid]     = weights[2 * tid];
            nodes[3 * tid + 1] = weights[2 * tid + 1];
            nodes[3 * tid + 2] = biases[tid];
        }
    }

    // ---- Load + convert the wave's 16x256 A panel into 8 bf16 fragments ----
    // A 16x32 layout (matches B): lane (g,n) holds row M=n, K = 32s + 16g + [0..15].
    v16bf afrag[8];
    {
        const float* xrow = x + (size_t)(row0 + n) * LCOLS;
        #pragma unroll
        for (int s = 0; s < 8; ++s) {
            const int k0 = 32 * s + 16 * g;
            const int k1 = k0 + 8;
            v4f p0 = __builtin_nontemporal_load((const v4f*)(xrow + k0));
            v4f p1 = __builtin_nontemporal_load((const v4f*)(xrow + k0 + 4));
            v4f p2 = __builtin_nontemporal_load((const v4f*)(xrow + k1));
            v4f p3 = __builtin_nontemporal_load((const v4f*)(xrow + k1 + 4));
            v16bf a;
            #pragma unroll
            for (int e = 0; e < 4; ++e) {
                a[e]      = (__bf16)p0[e];
                a[4 + e]  = (__bf16)p1[e];
                a[8 + e]  = (__bf16)p2[e];
                a[12 + e] = (__bf16)p3[e];
            }
            afrag[s] = a;
        }
    }

    __syncthreads();

    // Tree level offsets: level l has 128>>l pairs.
    const int loff[8] = {0, 128, 192, 224, 240, 248, 252, 254};

    float stage[8];

    #pragma unroll 1
    for (int t = 0; t < 16; ++t) {
        // ---- B fragments for this N-tile, all 8 K-steps, from LDS ----
        // B 32x16 bf16 layout: lane (g,n) holds column N=n (= W row 16t+n),
        // K = 32s + 16g + [0..15] -> 32 contiguous bytes of the LDS W row.
        const char* wrow = smem + (16 * t + n) * W_PITCH_B + 32 * g;
        v16bf bfrag[8];
        #pragma unroll
        for (int s = 0; s < 8; ++s)
            bfrag[s] = *(const v16bf*)(wrow + 64 * s);

        v8f acc = {};
        #pragma unroll
        for (int s = 0; s < 8; ++s)
            acc = __builtin_amdgcn_wmma_f32_16x16x32_bf16(
                false, afrag[s], false, bfrag[s], (short)0, acc, false, false);

        // Leaf activation. Lane holds column c for rows r + 8g (r = VGPR index).
        float v[8];
        #pragma unroll
        for (int r = 0; r < 8; ++r) v[r] = sigmoidf_(acc[r] + BIAS_SHIFT_V);

        // ---- Tree levels 0..3: butterflies over columns within the tile ----
        const int c = 16 * t + n;
        #pragma unroll
        for (int l = 0; l < 4; ++l) {
            const int q   = 3 * (loff[l] + (c >> (l + 1)));
            const float w0  = nodes[q];
            const float w1  = nodes[q + 1];
            const float lam = sigmoidf_(nodes[q + 2]);
            const bool left = ((c >> l) & 1) == 0;
            #pragma unroll
            for (int r = 0; r < 8; ++r) {
                float other = __shfl_xor(v[r], 1 << l, 32);
                float lv = (left ? v[r] : other) * w0;
                float rv = (left ? other : v[r]) * w1;
                v[r] = soft_gcd(lv, rv, lam);
            }
        }
        // Butterfly => every lane holds tile t's value; keep it in lane n==t.
        #pragma unroll
        for (int r = 0; r < 8; ++r)
            if (n == t) stage[r] = v[r];
    }

    // ---- Tree levels 4..7: butterflies across tiles (lane index = tile) ----
    #pragma unroll
    for (int l = 0; l < 4; ++l) {
        const int q   = 3 * (loff[4 + l] + (n >> (l + 1)));
        const float w0  = nodes[q];
        const float w1  = nodes[q + 1];
        const float lam = sigmoidf_(nodes[q + 2]);
        const bool left = ((n >> l) & 1) == 0;
        #pragma unroll
        for (int r = 0; r < 8; ++r) {
            float other = __shfl_xor(stage[r], 1 << l, 32);
            float lv = (left ? stage[r] : other) * w0;
            float rv = (left ? other : stage[r]) * w1;
            stage[r] = soft_gcd(lv, rv, lam);
        }
    }

    // ---- Output head: 16 scalars per wave ----
    const float wo = w_out[0], bo = b_out[0];
    if (n == 0) {
        #pragma unroll
        for (int r = 0; r < 8; ++r)
            out[row0 + 8 * g + r] = sigmoidf_(stage[r] * wo + bo);
    }
}

extern "C" void kernel_launch(void* const* d_in, const int* in_sizes, int n_in,
                              void* d_out, int out_size, void* d_ws, size_t ws_size,
                              hipStream_t stream) {
    const float* x       = (const float*)d_in[0];
    const float* W_leaf  = (const float*)d_in[1];
    const float* weights = (const float*)d_in[2];
    const float* biases  = (const float*)d_in[3];
    const float* w_out   = (const float*)d_in[4];
    const float* b_out   = (const float*)d_in[5];
    float* out           = (float*)d_out;

    __bf16* Wb = (__bf16*)d_ws;   // 256*256*2 = 128 KB of scratch

    const int rows = in_sizes[0] / LCOLS;   // 65536

    // 1) Convert W_leaf to bf16 once (tiny: 64K elements).
    hipLaunchKernelGGL(w_to_bf16, dim3((LCOLS * LCOLS) / 256), dim3(256), 0, stream,
                       W_leaf, Wb);

    // 2) Fused GEMM + tree + head. 8 waves/block, 16 rows/wave => 128 rows/block.
    //    ~132 KB dynamic LDS per block (CDNA5 WGP has 320 KB) => 2 blocks/WGP.
    hipLaunchKernelGGL(tree_logic_net, dim3(rows / 128), dim3(256), SMEM_BYTES, stream,
                       x, Wb, weights, biases, w_out, b_out, out);
}
